// CorrespondenceAttn_2199023256143
// MI455X (gfx1250) — matrix operand order repaired and checked by reading, not compiled, not run on hardware
//
#include <hip/hip_runtime.h>

#define C_DIM 320
#define NQ    2048          // b*m*h*w
#define SNUM  8
#define NTOK  72            // S*9
#define HEADS 10
#define DH    32

typedef _Float16 f16;
typedef _Float16 f16x16 __attribute__((ext_vector_type(16)));
typedef _Float16 f16x8  __attribute__((ext_vector_type(8)));
typedef float    f32x8  __attribute__((ext_vector_type(8)));

union Frag16 { f16x16 v; f16x8 h[2]; };

// ------------------------------------------------------------------
// Geometry: sel (stable partition of degree mask), homo_l, homo_r
// ------------------------------------------------------------------
__device__ inline void mul3(const float* a, const float* b, float* o) {
  for (int i = 0; i < 3; ++i)
    for (int j = 0; j < 3; ++j)
      o[i*3+j] = a[i*3+0]*b[0*3+j] + a[i*3+1]*b[1*3+j] + a[i*3+2]*b[2*3+j];
}
__device__ inline void inv3(const float* a, float* o) {
  float c00 = a[4]*a[8] - a[5]*a[7];
  float c01 = a[5]*a[6] - a[3]*a[8];
  float c02 = a[3]*a[7] - a[4]*a[6];
  float det = a[0]*c00 + a[1]*c01 + a[2]*c02;
  float id  = 1.0f / det;
  o[0] = c00*id;                  o[1] = (a[2]*a[7]-a[1]*a[8])*id; o[2] = (a[1]*a[5]-a[2]*a[4])*id;
  o[3] = c01*id;                  o[4] = (a[0]*a[8]-a[2]*a[6])*id; o[5] = (a[2]*a[3]-a[0]*a[5])*id;
  o[6] = c02*id;                  o[7] = (a[1]*a[6]-a[0]*a[7])*id; o[8] = (a[0]*a[4]-a[1]*a[3])*id;
}

__global__ void k_geom(const float* __restrict__ R_hr, const float* __restrict__ K_hr,
                       const float* __restrict__ R_lr, const float* __restrict__ K_lr,
                       const float* __restrict__ deg_hr, const float* __restrict__ deg_lr,
                       float* __restrict__ homoL, float* __restrict__ homoR,
                       int* __restrict__ sel) {
  int tid = threadIdx.x;
  if (tid < 2) {                     // stable argsort(!mask): true views first, in order
    int mi = tid, cnt = 0;
    for (int j = 0; j < SNUM; ++j) {
      float d = deg_hr[mi] - deg_lr[j];
      if (fabsf(d) < 90.f || fabsf(d - 360.f) < 90.f) sel[mi*SNUM + cnt++] = j;
    }
    for (int j = 0; j < SNUM; ++j) {
      float d = deg_hr[mi] - deg_lr[j];
      if (!(fabsf(d) < 90.f || fabsf(d - 360.f) < 90.f)) sel[mi*SNUM + cnt++] = j;
    }
  }
  __syncthreads();
  if (tid < 2*SNUM) {
    int mi = tid / SNUM, s = tid % SNUM;
    int ss = sel[mi*SNUM + s];
    float iR[9], iK[9], t0[9], t1[9], hl[9];
    inv3(R_lr + ss*9, iR);
    inv3(K_hr + mi*9, iK);
    mul3(K_lr + ss*9, iR, t0);       // K_r @ inv(R_r)
    mul3(t0, R_hr + mi*9, t1);       // @ R_hr
    mul3(t1, iK, hl);                // @ inv(K_hr)
    for (int i = 0; i < 9; ++i) homoL[tid*9 + i] = hl[i];
    float hr[9]; inv3(hl, hr);
    for (int i = 0; i < 9; ++i) homoR[tid*9 + i] = hr[i];
  }
}

// ------------------------------------------------------------------
// Weight convert: f32 [K][N] -> f16 transposed [N][K] (WMMA B layout)
// ------------------------------------------------------------------
__global__ void k_convert(const float* __restrict__ src, f16* __restrict__ dst, int K, int N) {
  int i = blockIdx.x * blockDim.x + threadIdx.x;
  if (i >= K * N) return;
  int k = i / N, n = i % N;
  dst[(size_t)n * K + k] = (f16)src[i];
}

// ------------------------------------------------------------------
// Gather x_hr (bm,c,h,w) -> x0 (q, c) row-major
// ------------------------------------------------------------------
__global__ void k_prep(const float* __restrict__ x_hr, float* __restrict__ x0) {
  int i = blockIdx.x * blockDim.x + threadIdx.x;
  if (i >= NQ * C_DIM) return;
  int q = i / C_DIM, c = i % C_DIM;
  int bm = q >> 10, pix = q & 1023;
  x0[i] = x_hr[((size_t)bm * C_DIM + c) * 1024 + pix];
}

// ------------------------------------------------------------------
// x = resid + gin + bias ; a = LN(x)*g+b (+qpe) -> f16
// ------------------------------------------------------------------
__global__ void k_ln(const float* __restrict__ resid, const float* __restrict__ gin,
                     const float* __restrict__ bias, float* __restrict__ xout,
                     f16* __restrict__ aout, const float* __restrict__ g,
                     const float* __restrict__ b, int addqpe) {
  int q = blockIdx.x, tid = threadIdx.x;
  __shared__ float xs[C_DIM];
  __shared__ float rs[256], rs2[256];
  for (int c = tid; c < C_DIM; c += 256) {
    float v = (resid ? resid[(size_t)q*C_DIM + c] : 0.f)
            + (gin   ? gin[(size_t)q*C_DIM + c]   : 0.f)
            + (bias  ? bias[c] : 0.f);
    xs[c] = v;
    if (xout) xout[(size_t)q*C_DIM + c] = v;
  }
  __syncthreads();
  float s = 0.f, s2 = 0.f;
  for (int c = tid; c < C_DIM; c += 256) { float v = xs[c]; s += v; s2 += v*v; }
  rs[tid] = s; rs2[tid] = s2; __syncthreads();
  for (int off = 128; off > 0; off >>= 1) {
    if (tid < off) { rs[tid] += rs[tid+off]; rs2[tid] += rs2[tid+off]; }
    __syncthreads();
  }
  float mu  = rs[0]  * (1.0f / C_DIM);
  float var = rs2[0] * (1.0f / C_DIM) - mu*mu;
  float inv = rsqrtf(var + 1e-5f);
  for (int c = tid; c < C_DIM; c += 256) {
    float a = (xs[c] - mu) * inv * g[c] + b[c];
    if (addqpe) a += ((c & 3) >= 2) ? 1.0f : 0.0f;   // pe(0): [sin0,sin0,cos0,cos0]
    aout[(size_t)q*C_DIM + c] = (f16)a;
  }
}

// ------------------------------------------------------------------
// WMMA GEMM, register-blocked: C(MxN,f32) = A(MxK,f16) @ Bt(NxK,f16)^T
// block = 128 (4 waves); each wave owns 5 N-tiles (A-frag reused x5);
// one block covers 320 columns. grid = (M/16, N/320)
// ------------------------------------------------------------------
#define NTILES_PER_WAVE 5

__global__ void k_gemm(const f16* __restrict__ A, const f16* __restrict__ Bt,
                       float* __restrict__ C, f16* __restrict__ Cf16,
                       const float* __restrict__ bias, int M, int N, int K) {
  int wave = threadIdx.x >> 5, lane = threadIdx.x & 31;
  int nt0  = blockIdx.y * 20 + wave * NTILES_PER_WAVE;
  int mr0  = blockIdx.x * 16;
  int arow = mr0 + (lane & 15);

  // prefetch the B panel for this wave (global_prefetch_b8)
  __builtin_prefetch(Bt + (size_t)(nt0 * 16) * K, 0, 0);

  f32x8 acc[NTILES_PER_WAVE] = {};
  for (int ks = 0; ks < (K >> 5); ++ks) {
    Frag16 a;
    int kb  = ks*32 + ((lane & 16) ? 16 : 0);
    a.h[0] = *(const f16x8*)(A + (size_t)arow * K + kb);
    a.h[1] = *(const f16x8*)(A + (size_t)arow * K + kb + 8);
    int kb2 = ks*32 + ((lane & 16) ? 16 : 0);
#pragma unroll
    for (int j = 0; j < NTILES_PER_WAVE; ++j) {
      int ncol = (nt0 + j) * 16 + (lane & 15);
      Frag16 b;
      b.h[0] = *(const f16x8*)(Bt + (size_t)ncol * K + kb2);
      b.h[1] = *(const f16x8*)(Bt + (size_t)ncol * K + kb2 + 8);
      acc[j] = __builtin_amdgcn_wmma_f32_16x16x32_f16(false, a.v, false, b.v,
                                                      (short)0, acc[j], false, false);
    }
  }
#pragma unroll
  for (int j = 0; j < NTILES_PER_WAVE; ++j) {
    int ncol = (nt0 + j) * 16 + (lane & 15);
    float bv = bias ? bias[ncol] : 0.f;
    for (int r = 0; r < 8; ++r) {
      int mr = mr0 + ((lane < 16) ? r : 8 + r);
      float vv = acc[j][r] + bv;
      if (C)    C[(size_t)mr * N + ncol] = vv;
      if (Cf16) Cf16[(size_t)mr * N + ncol] = (f16)vv;
    }
  }
}

// ------------------------------------------------------------------
// Fused correspondence cross-attention. One block (256 thr) per query.
// Builds 72 warped+PE tokens in LDS, then logits = T @ (Wk2 . q_h),
// softmax, u = probs^T @ T, o = u_h @ Wv2[:,h-block] -> f16.
// ------------------------------------------------------------------
#define SM_TT     0                      // f16 [96][320]
#define SM_W2Q    61440                  // f16 [16][320]
#define SM_PROBS  71680                  // f16 [16][96]
#define SM_LOGITS 74752                  // f32 [80][16]
#define SM_U      79872                  // f32 [16][320]
#define SM_TOKX   100352
#define SM_TOKY   100640
#define SM_TOKM   100928
#define SM_FEAT   101216
#define SM_ATTN_BYTES 101504

__global__ void k_attn(const float* __restrict__ x_lr, const float* __restrict__ Wk2,
                       const float* __restrict__ Wv2, const float* __restrict__ q2,
                       const float* __restrict__ homoL, const float* __restrict__ homoR,
                       const int* __restrict__ sel, f16* __restrict__ oOut) {
  extern __shared__ char smem[];
  f16*   Tt      = (f16*)(smem + SM_TT);
  f16*   w2qT    = (f16*)(smem + SM_W2Q);
  f16*   probsA  = (f16*)(smem + SM_PROBS);
  float* logitsS = (float*)(smem + SM_LOGITS);
  float* uS      = (float*)(smem + SM_U);
  float* tokX    = (float*)(smem + SM_TOKX);
  float* tokY    = (float*)(smem + SM_TOKY);
  float* tokM    = (float*)(smem + SM_TOKM);
  int*   featB   = (int*)(smem + SM_FEAT);

  int q = blockIdx.x, tid = threadIdx.x;
  int mi = q >> 10, pix = q & 1023;
  int py = pix >> 5, px = pix & 31;

  // ---- Phase A: per-token geometry ----
  if (tid < NTOK) {
    int t = tid, s = t / 9, kk = t % 9;
    int ss = sel[mi*SNUM + s];
    const float* hl = homoL + (mi*SNUM + s)*9;
    const float* hr = homoR + (mi*SNUM + s)*9;
    float fx = 4.0f + 8.0f * px, fy = 4.0f + 8.0f * py;   // hr-image query center
    float wl = hl[6]*fx + hl[7]*fy + hl[8];
    float ux = (hl[0]*fx + hl[1]*fy + hl[2]) / wl;
    float uy = (hl[3]*fx + hl[4]*fy + hl[5]) / wl;
    float ox = ux * 0.125f + (float)(kk/3 - 1);           // xy_q + offset (x)
    float oy = uy * 0.125f + (float)(kk%3 - 1);           // (y)
    float rxf = rintf((ox + 0.5f) * 8.0f - 0.5f);
    float ryf = rintf((oy + 0.5f) * 8.0f - 0.5f);
    float mk = (rxf >= 0.f && rxf < 128.f && ryf >= 0.f && ryf < 128.f) ? 1.f : 0.f;
    float cx = fminf(fmaxf(rxf, 0.f), 127.f);
    float cy = fminf(fmaxf(ryf, 0.f), 127.f);
    float wr = hr[6]*cx + hr[7]*cy + hr[8];
    float prx = (hr[0]*cx + hr[1]*cy + hr[2]) / wr;       // project back to hr image
    float pry = (hr[3]*cx + hr[4]*cy + hr[5]) / wr;
    tokX[t] = (prx - fx) * 0.125f;
    tokY[t] = (pry - fy) * 0.125f;
    tokM[t] = mk;
    int rsx = min(max((int)rintf(ox), 0), 15);
    int rsy = min(max((int)rintf(oy), 0), 15);
    featB[t] = ss * C_DIM * 256 + rsy * 16 + rsx;
  }
  __syncthreads();

  // ---- Phase B: build tokens (gather + NeRF PE), rows 72..95 zero-padded ----
  for (int i = tid; i < 96 * C_DIM; i += 256) {
    int t = i / C_DIM, c = i % C_DIM;
    f16 val = (f16)0;
    if (t < NTOK) {
      float feat = x_lr[featB[t] + c * 256];
      int md = c & 3;
      float ang = ldexpf((md & 1) ? tokY[t] : tokX[t], c >> 2);   // x * 2^f
      float p = (md < 2) ? sinf(ang) : cosf(ang);
      val = (f16)((feat + p) * tokM[t]);
    }
    Tt[i] = val;
  }
  __syncthreads();

  // ---- Phase C: w2qT[h][c] = scale * sum_d Wk2[c][h*32+d] * q2[q][h*32+d] ----
  {
    const float* qq = q2 + (size_t)q * C_DIM;
    for (int i = tid; i < 16 * C_DIM; i += 256) {
      int h = i / C_DIM, c = i % C_DIM;
      float acc = 0.f;
      if (h < HEADS) {
        const float* wk = Wk2 + (size_t)c * C_DIM + h * DH;
        for (int d = 0; d < DH; ++d) acc += wk[d] * qq[h*DH + d];
        acc *= 0.17677669529663689f;   // 32^-0.5
      }
      w2qT[i] = (f16)acc;
    }
  }
  __syncthreads();

  int wave = tid >> 5, lane = tid & 31;

  // ---- Phase D: logits (80x16) = T(80x320) @ w2q(320x16) via WMMA ----
  if (wave < 5) {
    f32x8 acc = {};
    int arow = wave*16 + (lane & 15);
    for (int ks = 0; ks < 10; ++ks) {
      Frag16 a, b;
      int kb  = ks*32 + ((lane & 16) ? 16 : 0);
      a.h[0] = *(const f16x8*)(Tt + arow*C_DIM + kb);
      a.h[1] = *(const f16x8*)(Tt + arow*C_DIM + kb + 8);
      int kb2 = ks*32 + ((lane & 16) ? 16 : 0);
      b.h[0] = *(const f16x8*)(w2qT + (lane & 15)*C_DIM + kb2);
      b.h[1] = *(const f16x8*)(w2qT + (lane & 15)*C_DIM + kb2 + 8);
      acc = __builtin_amdgcn_wmma_f32_16x16x32_f16(false, a.v, false, b.v,
                                                   (short)0, acc, false, false);
    }
    for (int r = 0; r < 8; ++r)
      logitsS[(wave*16 + ((lane < 16) ? r : 8 + r)) * 16 + (lane & 15)] = acc[r];
  }
  __syncthreads();

  // ---- Phase E: per-head softmax over 72 tokens (pad -> 0 weight) ----
  if (tid < 16) {
    int h = tid;
    if (h < HEADS) {
      float mx = -3.0e38f;
      for (int t = 0; t < NTOK; ++t) mx = fmaxf(mx, logitsS[t*16 + h]);
      float sm = 0.f;
      for (int t = 0; t < NTOK; ++t) sm += expf(logitsS[t*16 + h] - mx);
      float inv = 1.0f / sm;
      for (int t = 0; t < 96; ++t)
        probsA[h*96 + t] = (t < NTOK) ? (f16)(expf(logitsS[t*16 + h] - mx) * inv) : (f16)0;
    } else {
      for (int t = 0; t < 96; ++t) probsA[h*96 + t] = (f16)0;
    }
  }
  __syncthreads();

  // ---- Phase F: u(16x320) = probs(16x96) @ T(96x320) via WMMA ----
  for (int nt = wave; nt < 20; nt += 8) {
    f32x8 acc = {};
    int arow = lane & 15;
    for (int ks = 0; ks < 3; ++ks) {
      Frag16 a, b;
      int kb  = ks*32 + ((lane & 16) ? 16 : 0);
      a.h[0] = *(const f16x8*)(probsA + arow*96 + kb);
      a.h[1] = *(const f16x8*)(probsA + arow*96 + kb + 8);
      int kb2 = ks*32 + ((lane & 16) ? 16 : 0);
      int bcol = nt*16 + (lane & 15);
      for (int e = 0; e < 16; ++e) b.v[e] = Tt[(kb2 + e)*C_DIM + bcol];
      acc = __builtin_amdgcn_wmma_f32_16x16x32_f16(false, a.v, false, b.v,
                                                   (short)0, acc, false, false);
    }
    for (int r = 0; r < 8; ++r)
      uS[((lane < 16) ? r : 8 + r) * C_DIM + nt*16 + (lane & 15)] = acc[r];
  }
  __syncthreads();

  // ---- Phase G: o[h*32+d] = sum_c u[h][c] * Wv2[c][h*32+d] ----
  for (int co = tid; co < C_DIM; co += 256) {
    int h = co >> 5;
    float acc = 0.f;
    for (int c = 0; c < C_DIM; ++c) acc += uS[h*C_DIM + c] * Wv2[(size_t)c*C_DIM + co];
    oOut[(size_t)q*C_DIM + co] = (f16)acc;
  }
}

// ------------------------------------------------------------------
// GEGLU: gg = (hpre[:, :1280]+bf1_a) * gelu_exact(hpre[:, 1280:]+bf1_g) -> f16
// ------------------------------------------------------------------
__global__ void k_geglu(const float* __restrict__ hpre, const float* __restrict__ bf1,
                        f16* __restrict__ gg) {
  int i = blockIdx.x * blockDim.x + threadIdx.x;
  if (i >= NQ * 1280) return;
  int q = i / 1280, j = i % 1280;
  float a  = hpre[(size_t)q*2560 + j] + bf1[j];
  float gx = hpre[(size_t)q*2560 + 1280 + j] + bf1[1280 + j];
  float gl = 0.5f * gx * (1.0f + erff(gx * 0.70710678118654752f));
  gg[i] = (f16)(a * gl);
}

// ------------------------------------------------------------------
// out(bm,c,h,w) = x2 + r3 + bf2
// ------------------------------------------------------------------
__global__ void k_out(const float* __restrict__ x2, const float* __restrict__ r3,
                      const float* __restrict__ bf2, float* __restrict__ out) {
  int i = blockIdx.x * blockDim.x + threadIdx.x;
  if (i >= NQ * C_DIM) return;
  int q = i / C_DIM, c = i % C_DIM;
  int bm = q >> 10, pix = q & 1023;
  out[((size_t)bm * C_DIM + c) * 1024 + pix] = x2[i] + r3[i] + bf2[c];
}

// ------------------------------------------------------------------
extern "C" void kernel_launch(void* const* d_in, const int* in_sizes, int n_in,
                              void* d_out, int out_size, void* d_ws, size_t ws_size,
                              hipStream_t stream) {
  (void)in_sizes; (void)n_in; (void)out_size; (void)ws_size;
  const float* x_hr  = (const float*)d_in[0];
  const float* x_lr  = (const float*)d_in[1];
  const float* R_hr  = (const float*)d_in[2];
  const float* K_hr  = (const float*)d_in[3];
  const float* R_lr  = (const float*)d_in[4];
  const float* K_lr  = (const float*)d_in[5];
  const float* degh  = (const float*)d_in[6];
  const float* degl  = (const float*)d_in[7];
  const float* n1_g = (const float*)d_in[11]; const float* n1_b = (const float*)d_in[12];
  const float* Wv1  = (const float*)d_in[15]; const float* Wo1  = (const float*)d_in[16];
  const float* bo1  = (const float*)d_in[17];
  const float* n2_g = (const float*)d_in[18]; const float* n2_b = (const float*)d_in[19];
  const float* Wq2  = (const float*)d_in[20]; const float* Wk2  = (const float*)d_in[21];
  const float* Wv2  = (const float*)d_in[22]; const float* Wo2  = (const float*)d_in[23];
  const float* bo2  = (const float*)d_in[24];
  const float* n3_g = (const float*)d_in[25]; const float* n3_b = (const float*)d_in[26];
  const float* Wf1  = (const float*)d_in[27]; const float* bf1  = (const float*)d_in[28];
  const float* Wf2  = (const float*)d_in[29]; const float* bf2  = (const float*)d_in[30];

  char* p = (char*)d_ws;
  auto carve = [&](size_t bytes) { void* r = (void*)p; p += (bytes + 255) & ~(size_t)255; return r; };
  float* homoL = (float*)carve(16*9*4);
  float* homoR = (float*)carve(16*9*4);
  int*   sel   = (int*)  carve(16*4);
  f16* Wv1t = (f16*)carve((size_t)320*320*2);
  f16* Wo1t = (f16*)carve((size_t)320*320*2);
  f16* Wq2t = (f16*)carve((size_t)320*320*2);
  f16* Wo2t = (f16*)carve((size_t)320*320*2);
  f16* Wf1t = (f16*)carve((size_t)320*2560*2);
  f16* Wf2t = (f16*)carve((size_t)1280*320*2);
  float* x0   = (float*)carve((size_t)NQ*C_DIM*4);
  f16*   a1   = (f16*)  carve((size_t)NQ*C_DIM*2);
  f16*   t1   = (f16*)  carve((size_t)NQ*C_DIM*2);
  float* r1   = (float*)carve((size_t)NQ*C_DIM*4);
  float* x1   = (float*)carve((size_t)NQ*C_DIM*4);
  f16*   a2   = (f16*)  carve((size_t)NQ*C_DIM*2);
  float* q2w  = (float*)carve((size_t)NQ*C_DIM*4);
  f16*   oO   = (f16*)  carve((size_t)NQ*C_DIM*2);
  float* r2   = (float*)carve((size_t)NQ*C_DIM*4);
  float* x2   = (float*)carve((size_t)NQ*C_DIM*4);
  f16*   a3   = (f16*)  carve((size_t)NQ*C_DIM*2);
  float* hpre = (float*)carve((size_t)NQ*2560*4);
  f16*   gg   = (f16*)  carve((size_t)NQ*1280*2);
  float* r3   = (float*)carve((size_t)NQ*C_DIM*4);

  k_geom<<<1, 64, 0, stream>>>(R_hr, K_hr, R_lr, K_lr, degh, degl, homoL, homoR, sel);

  k_convert<<<400, 256, 0, stream>>>(Wv1, Wv1t, 320, 320);
  k_convert<<<400, 256, 0, stream>>>(Wo1, Wo1t, 320, 320);
  k_convert<<<400, 256, 0, stream>>>(Wq2, Wq2t, 320, 320);
  k_convert<<<400, 256, 0, stream>>>(Wo2, Wo2t, 320, 320);
  k_convert<<<3200, 256, 0, stream>>>(Wf1, Wf1t, 320, 2560);
  k_convert<<<1600, 256, 0, stream>>>(Wf2, Wf2t, 1280, 320);

  k_prep<<<2560, 256, 0, stream>>>(x_hr, x0);
  // a1 = LN1(x0) + qpe
  k_ln<<<NQ, 256, 0, stream>>>(nullptr, x0, nullptr, nullptr, a1, n1_g, n1_b, 1);
  // self-attn (len-1): x1 = x0 + a1 @ Wv1 @ Wo1 + bo1 ; a2 = LN2(x1) + qpe
  k_gemm<<<dim3(128, 1), 128, 0, stream>>>(a1, Wv1t, nullptr, t1, nullptr, NQ, 320, 320);
  k_gemm<<<dim3(128, 1), 128, 0, stream>>>(t1, Wo1t, r1, nullptr, nullptr, NQ, 320, 320);
  k_ln<<<NQ, 256, 0, stream>>>(x0, r1, bo1, x1, a2, n2_g, n2_b, 1);
  // q2 = a2 @ Wq2
  k_gemm<<<dim3(128, 1), 128, 0, stream>>>(a2, Wq2t, q2w, nullptr, nullptr, NQ, 320, 320);
  // fused warp-gather + PE + cross-attention
  k_attn<<<NQ, 256, SM_ATTN_BYTES, stream>>>(x_lr, Wk2, Wv2, q2w, homoL, homoR, sel, oO);
  // x2 = x1 + o @ Wo2 + bo2 ; a3 = LN3(x2)
  k_gemm<<<dim3(128, 1), 128, 0, stream>>>(oO, Wo2t, r2, nullptr, nullptr, NQ, 320, 320);
  k_ln<<<NQ, 256, 0, stream>>>(x1, r2, bo2, x2, a3, n3_g, n3_b, 0);
  // GEGLU FF
  k_gemm<<<dim3(128, 8), 128, 0, stream>>>(a3, Wf1t, hpre, nullptr, nullptr, NQ, 2560, 320);
  k_geglu<<<(NQ*1280)/256, 256, 0, stream>>>(hpre, bf1, gg);
  k_gemm<<<dim3(128, 1), 128, 0, stream>>>(gg, Wf2t, r3, nullptr, nullptr, NQ, 320, 1280);
  k_out<<<2560, 256, 0, stream>>>(x2, r3, bf2, (float*)d_out);
}
